// SlidingWindowTransformerModel_72533407694994
// MI455X (gfx1250) — matrix operand order repaired and checked by reading, not compiled, not run on hardware
//
#include <hip/hip_runtime.h>
#include <hip/hip_bf16.h>
#include <math.h>

// ---------------------------------------------------------------------------
// MI455X (gfx1250, wave32) implementation.
// Pipeline: segment-mean pool -> f16 WMMA GEMMs (16x64 tile per wave, A-frag
// reused across 4 N-tiles) -> persistent-WG GRU scans (WMMA recurrent matmul,
// gates in registers) -> WMMA up-proj + residual -> LayerNorm + head.
// ---------------------------------------------------------------------------

typedef __attribute__((ext_vector_type(16))) _Float16 v16h;
typedef __attribute__((ext_vector_type(8)))  _Float16 v8h;
typedef __attribute__((ext_vector_type(8)))  float    v8f;

#define B_   16
#define L_   2048
#define D_   1024
#define S_   64
#define H_   512
#define G3H  1536   // 3*H
#define ROWS 1024   // S_*B_

// ---- WMMA fragment loaders -------------------------------------------------
// A-matrix 16x32 f16: lane (g = lane>>4, m = lane&15) holds
//   K = g*16 .. g*16+15 (contig, v16h elems 0..15)
__device__ __forceinline__ v16h ldfragA(const _Float16* p) {
    v8h lo = *(const v8h*)(p);
    v8h hi = *(const v8h*)(p + 8);
    v16h r;
#pragma unroll
    for (int i = 0; i < 8; ++i) { r[i] = lo[i]; r[8 + i] = hi[i]; }
    return r;
}
// B-matrix 32x16 f16: lane (g, n = lane&15) holds K = g*16 .. g*16+15 (contig)
__device__ __forceinline__ v16h ldfragB(const _Float16* p) {
    v8h lo = *(const v8h*)(p);
    v8h hi = *(const v8h*)(p + 8);
    v16h r;
#pragma unroll
    for (int i = 0; i < 8; ++i) { r[i] = lo[i]; r[8 + i] = hi[i]; }
    return r;
}

__device__ __forceinline__ v8f wmma_f16(v16h a, v16h b, v8f c) {
    return __builtin_amdgcn_wmma_f32_16x16x32_f16(
        /*neg_a=*/false, a, /*neg_b=*/false, b,
        /*c_mod=*/(short)0, c, /*reuse_a=*/false, /*reuse_b=*/false);
}

// ---- fp32 -> fp16 weight conversion ---------------------------------------
__global__ __launch_bounds__(256) void cvt_f32_f16(const float* __restrict__ src,
                                                   _Float16* __restrict__ dst, int n) {
    int i = blockIdx.x * 256 + threadIdx.x;
    if (i < n) dst[i] = (_Float16)src[i];
}

// ---- ragged segment-mean pooling ------------------------------------------
// gather_indices sorted per row -> segment s of batch b is a contiguous token
// range found by binary search. One block per (s,b); row r = s*B + b (S,B,D).
__device__ __forceinline__ int lower_bound_i(const int* a, int n, int v) {
    int lo = 0, hi = n;
    while (lo < hi) { int mid = (lo + hi) >> 1; if (a[mid] < v) lo = mid + 1; else hi = mid; }
    return lo;
}

__global__ __launch_bounds__(256) void pool_kernel(const float* __restrict__ hs,
                                                   const int* __restrict__ gi,
                                                   float* __restrict__ pooledF,
                                                   _Float16* __restrict__ pooledH) {
    int r = blockIdx.x;            // r = s*16 + b
    int b = r & 15, s = r >> 4;
    __shared__ int bnd[2];
    if (threadIdx.x == 0) {
        const int* row = gi + (size_t)b * L_;
        bnd[0] = lower_bound_i(row, L_, s);
        bnd[1] = lower_bound_i(row, L_, s + 1);
    }
    __syncthreads();
    int lo = bnd[0], hi = bnd[1];
    float acc[4] = {0.f, 0.f, 0.f, 0.f};
    for (int t = lo; t < hi; ++t) {
        const float* p = hs + ((size_t)b * L_ + t) * D_;
#pragma unroll
        for (int q = 0; q < 4; ++q) acc[q] += p[threadIdx.x + q * 256];
    }
    float inv = 1.f / (float)(hi - lo);
#pragma unroll
    for (int q = 0; q < 4; ++q) {
        float v = acc[q] * inv;
        size_t idx = (size_t)r * D_ + threadIdx.x + q * 256;
        pooledF[idx] = v;
        pooledH[idx] = (_Float16)v;
    }
}

// ---- WMMA GEMM, 16x64 output per wave -------------------------------------
// Y[r,o] = sum_i A[r,i]*W[o,i] + bias[o] (+resid). A: MxK f16 row-major,
// W: NxK f16 row-major. Each wave owns 4 adjacent N-tiles so the A fragment
// is loaded once and fed to 4 WMMAs per k-step (4x A-traffic reduction).
__global__ __launch_bounds__(32) void gemm_wmma(const _Float16* __restrict__ A,
                                                const _Float16* __restrict__ W,
                                                const float* __restrict__ bias,
                                                const float* __restrict__ resid,
                                                _Float16* __restrict__ outH,
                                                float* __restrict__ outF,
                                                int M, int N, int K) {
    int lane = threadIdx.x;
    int g = lane >> 4, n = lane & 15;
    int m0 = blockIdx.y * 16, n0 = blockIdx.x * 64;

    v8f c[4];
#pragma unroll
    for (int t = 0; t < 4; ++t) {
        int col = n0 + t * 16 + n;
        float binit = bias ? bias[col] : 0.f;
#pragma unroll
        for (int j = 0; j < 8; ++j) {
            c[t][j] = binit;
            if (resid) c[t][j] += resid[(size_t)(m0 + g * 8 + j) * N + col];
        }
    }

    const _Float16* pa = A + (size_t)(m0 + n) * K + g * 16;   // A row = lane&15
    const _Float16* pb[4];
#pragma unroll
    for (int t = 0; t < 4; ++t)
        pb[t] = W + (size_t)(n0 + t * 16 + n) * K + g * 16;  // W row = out col

#pragma unroll 2
    for (int k = 0; k < K; k += 32) {
        if (k + 64 < K) {   // stream-ahead hint -> global_prefetch_b8
            __builtin_prefetch(pb[0] + k + 64, 0, 0);
            __builtin_prefetch(pb[2] + k + 64, 0, 0);
        }
        v16h af = ldfragA(pa + k);
#pragma unroll
        for (int t = 0; t < 4; ++t)
            c[t] = wmma_f16(af, ldfragB(pb[t] + k), c[t]);
    }

#pragma unroll
    for (int t = 0; t < 4; ++t) {
#pragma unroll
        for (int j = 0; j < 8; ++j) {
            size_t idx = (size_t)(m0 + g * 8 + j) * N + n0 + t * 16 + n;
            if (outF) outF[idx] = c[t][j];
            if (outH) outH[idx] = (_Float16)c[t][j];
        }
    }
}

// ---- persistent GRU scan ---------------------------------------------------
// One workgroup (32 waves). h (16xH) lives in LDS as f16 (WMMA A) + f32.
// Wave w owns features [16w,16w+16): computes r/z/n hW tiles via 3 WMMA
// accumulator chains sharing one A fragment, then the gate nonlinearity
// entirely in registers.
__global__ __launch_bounds__(1024) void gru_scan(const float* __restrict__ xW,  // (S*B, 3H)
                                                 const _Float16* __restrict__ Whh, // (3H, H)
                                                 const float* __restrict__ bhh,   // (3H)
                                                 _Float16* __restrict__ ysH) {    // (S*B, H)
    __shared__ _Float16 hH[16 * H_];
    __shared__ float    hF[16 * H_];
    int tid = threadIdx.x;
    int wave = tid >> 5;
    int lane = tid & 31;
    int g = lane >> 4, nn = lane & 15;
    int j0 = wave * 16;              // feature tile base for this wave

    for (int i = tid; i < 16 * H_; i += 1024) { hH[i] = (_Float16)0.f; hF[i] = 0.f; }
    __syncthreads();

    const _Float16* pa = &hH[nn * H_] + g * 16;
    const _Float16* pr = Whh + (size_t)(0 * H_ + j0 + nn) * H_ + g * 16;
    const _Float16* pz = Whh + (size_t)(1 * H_ + j0 + nn) * H_ + g * 16;
    const _Float16* pn = Whh + (size_t)(2 * H_ + j0 + nn) * H_ + g * 16;
    float br = bhh[0 * H_ + j0 + nn];
    float bz = bhh[1 * H_ + j0 + nn];
    float bn = bhh[2 * H_ + j0 + nn];

    for (int s = 0; s < S_; ++s) {
        v8f cr, cz, cn;
#pragma unroll
        for (int j = 0; j < 8; ++j) { cr[j] = br; cz[j] = bz; cn[j] = bn; }

#pragma unroll 4
        for (int k = 0; k < H_; k += 32) {
            v16h af = ldfragA(pa + k);   // h from LDS (ds_load_b128)
            cr = wmma_f16(af, ldfragB(pr + k), cr);
            cz = wmma_f16(af, ldfragB(pz + k), cz);
            cn = wmma_f16(af, ldfragB(pn + k), cn);
        }
        __syncthreads();  // all waves done reading old h

        const float* xrow = xW + (size_t)s * 16 * G3H;
#pragma unroll
        for (int j = 0; j < 8; ++j) {
            int m = g * 8 + j;           // batch index
            int f = j0 + nn;             // feature index
            float xr = xrow[(size_t)m * G3H + f];
            float xz = xrow[(size_t)m * G3H + H_ + f];
            float xn = xrow[(size_t)m * G3H + 2 * H_ + f];
            float rg = 1.f / (1.f + __expf(-(xr + cr[j])));
            float zg = 1.f / (1.f + __expf(-(xz + cz[j])));
            float ng = tanhf(xn + rg * cn[j]);
            float hold = hF[m * H_ + f];
            float hnew = (1.f - zg) * ng + zg * hold;
            hF[m * H_ + f] = hnew;
            hH[m * H_ + f] = (_Float16)hnew;
            ysH[((size_t)s * 16 + m) * H_ + f] = (_Float16)hnew;
        }
        __syncthreads();  // h updated before next step's WMMA reads
    }
}

// ---- LayerNorm over D + 3-class head --------------------------------------
__global__ __launch_bounds__(256) void ln_head(const float* __restrict__ x,
                                               const float* __restrict__ gamma,
                                               const float* __restrict__ beta,
                                               const float* __restrict__ Wh,  // (3, D) f32
                                               const float* __restrict__ bh,
                                               float* __restrict__ out) {
    int r = blockIdx.x;  // internal (s,b) order: r = s*16 + b
    __shared__ float red[256];
    const float* xr = x + (size_t)r * D_;
    float v[4], s1 = 0.f, s2 = 0.f;
#pragma unroll
    for (int q = 0; q < 4; ++q) {
        v[q] = xr[threadIdx.x + q * 256];
        s1 += v[q]; s2 += v[q] * v[q];
    }
    red[threadIdx.x] = s1; __syncthreads();
    for (int o = 128; o > 0; o >>= 1) { if (threadIdx.x < o) red[threadIdx.x] += red[threadIdx.x + o]; __syncthreads(); }
    float mu = red[0] / (float)D_; __syncthreads();
    red[threadIdx.x] = s2; __syncthreads();
    for (int o = 128; o > 0; o >>= 1) { if (threadIdx.x < o) red[threadIdx.x] += red[threadIdx.x + o]; __syncthreads(); }
    float var = red[0] / (float)D_ - mu * mu; __syncthreads();
    float rstd = rsqrtf(var + 1e-5f);

    float d[3] = {0.f, 0.f, 0.f};
#pragma unroll
    for (int q = 0; q < 4; ++q) {
        int c = threadIdx.x + q * 256;
        float y = (v[q] - mu) * rstd * gamma[c] + beta[c];
        d[0] += y * Wh[c];
        d[1] += y * Wh[D_ + c];
        d[2] += y * Wh[2 * D_ + c];
    }
    int b = r & 15, s = r >> 4;
    size_t orow = ((size_t)b * S_ + s) * 3;   // output is (B,S) order
    for (int cc = 0; cc < 3; ++cc) {
        red[threadIdx.x] = d[cc]; __syncthreads();
        for (int o = 128; o > 0; o >>= 1) { if (threadIdx.x < o) red[threadIdx.x] += red[threadIdx.x + o]; __syncthreads(); }
        if (threadIdx.x == 0) out[orow + cc] = red[0] + bh[cc];
        __syncthreads();
    }
}

// ---------------------------------------------------------------------------
extern "C" void kernel_launch(void* const* d_in, const int* in_sizes, int n_in,
                              void* d_out, int out_size, void* d_ws, size_t ws_size,
                              hipStream_t stream) {
    (void)in_sizes; (void)n_in; (void)out_size; (void)ws_size;

    const float* hs     = (const float*)d_in[0];
    const float* W_down = (const float*)d_in[1];
    const float* b_down = (const float*)d_in[2];
    const float* W_ih0  = (const float*)d_in[3];
    const float* W_hh0  = (const float*)d_in[4];
    const float* b_ih0  = (const float*)d_in[5];
    const float* b_hh0  = (const float*)d_in[6];
    const float* W_ih1  = (const float*)d_in[7];
    const float* W_hh1  = (const float*)d_in[8];
    const float* b_ih1  = (const float*)d_in[9];
    const float* b_hh1  = (const float*)d_in[10];
    const float* W_lin1 = (const float*)d_in[11];
    const float* b_lin1 = (const float*)d_in[12];
    const float* gamma  = (const float*)d_in[13];
    const float* beta   = (const float*)d_in[14];
    const float* W_head = (const float*)d_in[15];
    const float* b_head = (const float*)d_in[16];
    const int*   gi     = (const int*)d_in[17];
    float* out = (float*)d_out;

    // workspace layout (256B-aligned slices)
    char* ws = (char*)d_ws;
    size_t off = 0;
    auto carve = [&](size_t bytes) -> void* {
        void* p = ws + off; off += (bytes + 255) & ~(size_t)255; return p;
    };
    float*    pooledF = (float*)   carve((size_t)ROWS * D_ * 4);   // reused as x (residual out)
    _Float16* pooledH = (_Float16*)carve((size_t)ROWS * D_ * 2);
    _Float16* WdownH  = (_Float16*)carve((size_t)H_ * D_ * 2);
    _Float16* Wih0H   = (_Float16*)carve((size_t)G3H * H_ * 2);
    _Float16* Whh0H   = (_Float16*)carve((size_t)G3H * H_ * 2);
    _Float16* Wih1H   = (_Float16*)carve((size_t)G3H * H_ * 2);
    _Float16* Whh1H   = (_Float16*)carve((size_t)G3H * H_ * 2);
    _Float16* WlinH   = (_Float16*)carve((size_t)D_ * H_ * 2);
    _Float16* xdH     = (_Float16*)carve((size_t)ROWS * H_ * 2);
    float*    xWF     = (float*)   carve((size_t)ROWS * G3H * 4);  // reused by both layers
    _Float16* ys0H    = (_Float16*)carve((size_t)ROWS * H_ * 2);
    _Float16* ys1H    = (_Float16*)carve((size_t)ROWS * H_ * 2);

    // 1) weights -> f16
    auto cvt = [&](const float* s, _Float16* d, int n) {
        cvt_f32_f16<<<(n + 255) / 256, 256, 0, stream>>>(s, d, n);
    };
    cvt(W_down, WdownH, H_ * D_);
    cvt(W_ih0, Wih0H, G3H * H_);
    cvt(W_hh0, Whh0H, G3H * H_);
    cvt(W_ih1, Wih1H, G3H * H_);
    cvt(W_hh1, Whh1H, G3H * H_);
    cvt(W_lin1, WlinH, D_ * H_);

    // 2) segment-mean pooling (HBM-bound stage, ~134 MB streamed once)
    pool_kernel<<<ROWS, 256, 0, stream>>>(hs, gi, pooledF, pooledH);

    // 3) down-proj: (1024x1024)x(1024x512) -> xd f16
    gemm_wmma<<<dim3(H_ / 64, ROWS / 16), 32, 0, stream>>>(
        pooledH, WdownH, b_down, nullptr, xdH, nullptr, ROWS, H_, D_);

    // 4) layer-0 input proj: xW0 = xd @ W_ih0^T + b_ih0
    gemm_wmma<<<dim3(G3H / 64, ROWS / 16), 32, 0, stream>>>(
        xdH, Wih0H, b_ih0, nullptr, nullptr, xWF, ROWS, G3H, H_);

    // 5) layer-0 recurrent scan (persistent WG, WMMA on LDS-resident h)
    gru_scan<<<1, 1024, 0, stream>>>(xWF, Whh0H, b_hh0, ys0H);

    // 6) layer-1 input proj
    gemm_wmma<<<dim3(G3H / 64, ROWS / 16), 32, 0, stream>>>(
        ys0H, Wih1H, b_ih1, nullptr, nullptr, xWF, ROWS, G3H, H_);

    // 7) layer-1 recurrent scan
    gru_scan<<<1, 1024, 0, stream>>>(xWF, Whh1H, b_hh1, ys1H);

    // 8) up-proj + residual (in place over pooledF): x = pooled + ys1 @ W_lin1^T + b
    gemm_wmma<<<dim3(D_ / 64, ROWS / 16), 32, 0, stream>>>(
        ys1H, WlinH, b_lin1, pooledF, nullptr, pooledF, ROWS, D_, H_);

    // 9) LayerNorm + head, writing in (B,S) row order
    ln_head<<<ROWS, 256, 0, stream>>>(pooledF, gamma, beta, W_head, b_head, out);
}
